// MOELayer_1726576856632
// MI455X (gfx1250) — matrix operand order repaired and checked by reading, not compiled, not run on hardware
//
#include <hip/hip_runtime.h>

// ---------------------------------------------------------------------------
// MoE layer for MI455X (gfx1250): S=8192 M=1024 H=4096 E=8 topk=2 C=2048
// Expert GEMMs: v_wmma_f32_16x16x32_bf16, LDS double-buffered via
// global_load_async_to_lds_b128 (ASYNCcnt path).
// ---------------------------------------------------------------------------

typedef unsigned short u16;
typedef __attribute__((ext_vector_type(16))) __bf16 v16bf;
typedef __attribute__((ext_vector_type(8)))  float  v8f;

union Frag {
  v16bf v;
  uint4 q[2];
};

#define S_TOK 8192
#define MDIM  1024
#define HDIM  4096
#define NEXP  8
#define CAP   2048

static __device__ __forceinline__ u16 f2bf(float f) {
  unsigned u = __float_as_uint(f);
  unsigned r = u + 0x7fffu + ((u >> 16) & 1u);   // round-to-nearest-even
  return (u16)(r >> 16);
}

// ---------------------------------------------------------------------------
// Weight conversion: fp32 -> bf16, transposed so GEMM B-loads are K-contiguous
// w1 [E][M][H] -> w1t [E][H][M]   (GEMM1: K = M)
// w2 [E][H][M] -> w2t [E][M][H]   (GEMM2: K = H)
// ---------------------------------------------------------------------------
__global__ __launch_bounds__(256) void conv_w1_kernel(const float* __restrict__ w1,
                                                      u16* __restrict__ w1t) {
  size_t idx = (size_t)blockIdx.x * 256 + threadIdx.x;   // over E*H*M
  int m = (int)(idx & (MDIM - 1));
  size_t eh = idx >> 10;            // e*H + h
  int h = (int)(eh & (HDIM - 1));
  int e = (int)(eh >> 12);
  w1t[idx] = f2bf(w1[((size_t)(e * MDIM + m)) * HDIM + h]);
}

__global__ __launch_bounds__(256) void conv_w2_kernel(const float* __restrict__ w2,
                                                      u16* __restrict__ w2t) {
  size_t idx = (size_t)blockIdx.x * 256 + threadIdx.x;   // over E*M*H, (e*M+m)*H+h
  int h = (int)(idx & (HDIM - 1));
  size_t em = idx >> 12;            // e*M + m
  int m = (int)(em & (MDIM - 1));
  int e = (int)(em >> 10);
  w2t[idx] = f2bf(w2[((size_t)(e * HDIM + h)) * MDIM + m]);
}

// ---------------------------------------------------------------------------
// Gating: one wave32 per token. scores = x @ wg (8 dots of len 1024),
// top-2 of logits, gates = renormalized softmax over the top-2.
// ---------------------------------------------------------------------------
__global__ __launch_bounds__(256) void gate_kernel(const float* __restrict__ x,
                                                   const float* __restrict__ wg,
                                                   int* __restrict__ topk_idx,
                                                   float* __restrict__ gates) {
  int wave = threadIdx.x >> 5;
  int lane = threadIdx.x & 31;
  int s = blockIdx.x * 8 + wave;
  float acc[NEXP];
#pragma unroll
  for (int e = 0; e < NEXP; ++e) acc[e] = 0.f;
  const float* xr = x + (size_t)s * MDIM;
  for (int j = lane; j < MDIM; j += 32) {
    float xv = xr[j];
    const float* wr = wg + j * NEXP;
#pragma unroll
    for (int e = 0; e < NEXP; ++e) acc[e] += xv * wr[e];
  }
#pragma unroll
  for (int e = 0; e < NEXP; ++e)
    for (int off = 16; off; off >>= 1)
      acc[e] += __shfl_xor(acc[e], off, 32);
  if (lane == 0) {
    int i0 = 0; float v0 = acc[0];
#pragma unroll
    for (int e = 1; e < NEXP; ++e) if (acc[e] > v0) { v0 = acc[e]; i0 = e; }
    int i1 = -1; float v1 = -3.4e38f;
#pragma unroll
    for (int e = 0; e < NEXP; ++e) if (e != i0 && acc[e] > v1) { v1 = acc[e]; i1 = e; }
    float p1 = __expf(v1 - v0);           // p0 = 1
    float inv = 1.f / (1.f + p1);
    topk_idx[s * 2 + 0] = i0;
    topk_idx[s * 2 + 1] = i1;
    gates[s * 2 + 0] = inv;
    gates[s * 2 + 1] = p1 * inv;
  }
}

// ---------------------------------------------------------------------------
// Capacity scan: one wave per expert, ballot prefix over slots in j-major
// order (all j=0 in s-order, then j=1) == reference cumsum + offset.
// Writes flat_idx[s][j] and inverse map row_src[e*C+loc].
// ---------------------------------------------------------------------------
__global__ __launch_bounds__(256) void scan_kernel(const int* __restrict__ topk_idx,
                                                   int* __restrict__ flat_idx,
                                                   int* __restrict__ row_src) {
  const int e = threadIdx.x >> 5;
  const int lane = threadIdx.x & 31;
  int base = 0;
  for (int t = 0; t < 2 * S_TOK; t += 32) {
    int slot = t + lane;
    int j = slot >> 13;            // / S_TOK
    int s = slot & (S_TOK - 1);
    int idx = topk_idx[s * 2 + j];
    bool match = (idx == e);
    unsigned m = (unsigned)__ballot(match);
    int pref = __popc(m & ((1u << lane) - 1u));
    if (match) {
      int loc = base + pref;
      int flat = (loc < CAP) ? (e * CAP + loc) : (NEXP * CAP);
      flat_idx[s * 2 + j] = flat;
      if (loc < CAP) row_src[e * CAP + loc] = s;
    }
    base += __popc(m);
  }
  for (int loc = base + lane; loc < CAP; loc += 32)
    row_src[e * CAP + loc] = -1;
}

// ---------------------------------------------------------------------------
// Dispatch: disp[row][:] = bf16(x[row_src[row]][:]) or 0. One block per row.
// ---------------------------------------------------------------------------
__global__ __launch_bounds__(256) void disp_kernel(const float* __restrict__ x,
                                                   const int* __restrict__ row_src,
                                                   u16* __restrict__ disp) {
  int row = blockIdx.x;             // 0 .. E*C-1
  int src = row_src[row];
  int c = threadIdx.x * 4;
  uint2 out;
  if (src >= 0) {
    float4 v = *reinterpret_cast<const float4*>(x + (size_t)src * MDIM + c);
    out.x = (unsigned)f2bf(v.x) | ((unsigned)f2bf(v.y) << 16);
    out.y = (unsigned)f2bf(v.z) | ((unsigned)f2bf(v.w) << 16);
  } else {
    out.x = 0u; out.y = 0u;
  }
  *reinterpret_cast<uint2*>(disp + (size_t)row * MDIM + c) = out;
}

// ---------------------------------------------------------------------------
// Tiled WMMA GEMM:  Out[e] = A[e](2048 x K) * B[e](K x N)^T-layout + bias[e]
//   A row-major bf16 (lda=K), B stored N-major/K-contiguous bf16 (ldb=K).
//   Block tile 128x128, BK=32; LDS double-buffered, filled with
//   global_load_async_to_lds_b128 (4 per thread per tile, ASYNCcnt-tracked).
//   8 waves, each wave computes 32x64 via 2x4 v_wmma_f32_16x16x32_bf16.
//   RELU_BF16: fused bias+ReLU, bf16 store (h). else: bias, f32 store (eo).
// ---------------------------------------------------------------------------
template <bool RELU_BF16>
__global__ __launch_bounds__(256) void moe_gemm_kernel(const u16* __restrict__ A,
                                                       const u16* __restrict__ B,
                                                       const float* __restrict__ bias,
                                                       void* __restrict__ Out,
                                                       int N, int Kdim) {
  constexpr int BM = 128, BN = 128, BK = 32, LDT = 40;
  __shared__ u16 As[2][BM * LDT];
  __shared__ u16 Bs[2][BN * LDT];

  const int e = blockIdx.z;
  const int rowBase = blockIdx.x * BM;
  const int nBase = blockIdx.y * BN;
  const int tid = threadIdx.x;
  const int lane = tid & 31;
  const int wave = tid >> 5;
  const int wr = wave >> 1;        // 0..3  (row *32)
  const int wc = wave & 1;         // 0..1  (col *64)
  const int half = lane >> 4;
  const int l15 = lane & 15;

  const size_t Aexp = (size_t)(e * CAP + rowBase) * Kdim;
  const size_t Bexp = (size_t)(e * N + nBase) * Kdim;

  const int ldRow = tid >> 1;      // 0..127
  const int ldHalf = tid & 1;      // 16-element chunk within BK

  // Per-thread global base (element offset k added per tile).
  const u16* gA = A + Aexp + (size_t)ldRow * Kdim + ldHalf * 16;
  const u16* gB = B + Bexp + (size_t)ldRow * Kdim + ldHalf * 16;

  // 32-bit LDS byte addresses (generic-pointer low bits == LDS offset).
  unsigned ldsA[2], ldsB[2];
  ldsA[0] = (unsigned)(size_t)(void*)&As[0][ldRow * LDT + ldHalf * 16];
  ldsA[1] = (unsigned)(size_t)(void*)&As[1][ldRow * LDT + ldHalf * 16];
  ldsB[0] = (unsigned)(size_t)(void*)&Bs[0][ldRow * LDT + ldHalf * 16];
  ldsB[1] = (unsigned)(size_t)(void*)&Bs[1][ldRow * LDT + ldHalf * 16];

  // Async-copy one BK tile (32B per thread of A, 32B of B) into buffer `buf`.
  auto issue_tile = [&](int k0, int buf) {
    unsigned long long a0 = (unsigned long long)(size_t)(gA + k0);
    unsigned long long b0 = (unsigned long long)(size_t)(gB + k0);
    unsigned la = ldsA[buf], lb = ldsB[buf];
    // INST_OFFSET is added to both LDS and global addresses (ISA 10.x async).
    asm volatile("global_load_async_to_lds_b128 %0, %1, off"
                 :: "v"(la), "v"(a0) : "memory");
    asm volatile("global_load_async_to_lds_b128 %0, %1, off offset:16"
                 :: "v"(la), "v"(a0) : "memory");
    asm volatile("global_load_async_to_lds_b128 %0, %1, off"
                 :: "v"(lb), "v"(b0) : "memory");
    asm volatile("global_load_async_to_lds_b128 %0, %1, off offset:16"
                 :: "v"(lb), "v"(b0) : "memory");
  };

  v8f acc[2][4];
#pragma unroll
  for (int i = 0; i < 2; ++i)
#pragma unroll
    for (int j = 0; j < 4; ++j)
      acc[i][j] = v8f{0.f, 0.f, 0.f, 0.f, 0.f, 0.f, 0.f, 0.f};

  const int nK = Kdim / BK;
  issue_tile(0, 0);

  for (int kk = 0; kk < nK; ++kk) {
    const int buf = kk & 1;
    if (kk + 1 < nK) {
      issue_tile((kk + 1) * BK, (kk + 1) & 1);
      // 4 loads just issued in-flight; <=4 outstanding => tile kk complete.
      asm volatile("s_wait_asynccnt 0x4" ::: "memory");
    } else {
      asm volatile("s_wait_asynccnt 0x0" ::: "memory");
    }
    __syncthreads();   // all waves' async fills of buf are visible

    Frag aF[2];
    Frag bF[4];
#pragma unroll
    for (int ai = 0; ai < 2; ++ai) {
      int r = wr * 32 + ai * 16 + l15;   // lane holds A row M=l15
      // A frag: VGPR0-3 K = half*8..+7 ; VGPR4-7 K = 16+half*8..+7
      aF[ai].q[0] = *reinterpret_cast<const uint4*>(&As[buf][r * LDT + half * 8]);
      aF[ai].q[1] = *reinterpret_cast<const uint4*>(&As[buf][r * LDT + 16 + half * 8]);
    }
#pragma unroll
    for (int bi = 0; bi < 4; ++bi) {
      int c = wc * 64 + bi * 16 + l15;   // lane holds B column; K = half*8..+7, 16+half*8..+7
      bF[bi].q[0] = *reinterpret_cast<const uint4*>(&Bs[buf][c * LDT + half * 8]);
      bF[bi].q[1] = *reinterpret_cast<const uint4*>(&Bs[buf][c * LDT + 16 + half * 8]);
    }

#pragma unroll
    for (int ai = 0; ai < 2; ++ai)
#pragma unroll
      for (int bi = 0; bi < 4; ++bi)
        acc[ai][bi] = __builtin_amdgcn_wmma_f32_16x16x32_bf16(
            false, aF[ai].v, false, bF[bi].v, (short)0, acc[ai][bi],
            false, false);
    __syncthreads();   // buf may be overwritten by tile kk+2 next iteration
  }

  // Epilogue.  C/D layout: VGPR r -> M = r + 8*half, N = l15.
#pragma unroll
  for (int ai = 0; ai < 2; ++ai) {
#pragma unroll
    for (int bi = 0; bi < 4; ++bi) {
      int col = nBase + wc * 64 + bi * 16 + l15;
      float bv = bias[e * N + col];
#pragma unroll
      for (int r = 0; r < 8; ++r) {
        int row = rowBase + wr * 32 + ai * 16 + r + half * 8;
        float v = acc[ai][bi][r] + bv;
        size_t oidx = (size_t)(e * CAP + row) * N + col;
        if constexpr (RELU_BF16) {
          v = v > 0.f ? v : 0.f;
          ((u16*)Out)[oidx] = f2bf(v);
        } else {
          ((float*)Out)[oidx] = v;
        }
      }
    }
  }
}

// ---------------------------------------------------------------------------
// Combine: y[s] = sum_j gates[s][j] * eo[flat_idx[s][j]]  (sentinel -> 0)
// ---------------------------------------------------------------------------
__global__ __launch_bounds__(256) void combine_kernel(const float* __restrict__ eo,
                                                      const int* __restrict__ flat_idx,
                                                      const float* __restrict__ gates,
                                                      float* __restrict__ y) {
  int s = blockIdx.x;
  int c = threadIdx.x * 4;
  float4 a = {0.f, 0.f, 0.f, 0.f};
#pragma unroll
  for (int j = 0; j < 2; ++j) {
    int f = flat_idx[s * 2 + j];
    float g = gates[s * 2 + j];
    if (f < NEXP * CAP) {
      float4 v = *reinterpret_cast<const float4*>(eo + (size_t)f * MDIM + c);
      a.x += g * v.x; a.y += g * v.y; a.z += g * v.z; a.w += g * v.w;
    }
  }
  *reinterpret_cast<float4*>(y + (size_t)s * MDIM + c) = a;
}

// ---------------------------------------------------------------------------
extern "C" void kernel_launch(void* const* d_in, const int* in_sizes, int n_in,
                              void* d_out, int out_size, void* d_ws, size_t ws_size,
                              hipStream_t stream) {
  const float* x  = (const float*)d_in[0];
  const float* wg = (const float*)d_in[1];
  const float* w1 = (const float*)d_in[2];
  const float* b1 = (const float*)d_in[3];
  const float* w2 = (const float*)d_in[4];
  const float* b2 = (const float*)d_in[5];

  char* ws = (char*)d_ws;
  // Workspace layout (bytes), all offsets 256B-aligned:
  u16*   w1t  = (u16*)(ws + 0);                      //  64 MB  [E][H][M] bf16
  u16*   w2t  = (u16*)(ws + 67108864);               //  64 MB  [E][M][H] bf16
  u16*   disp = (u16*)(ws + 134217728);              //  32 MB  [E*C][M] bf16
  u16*   hbuf = (u16*)(ws + 167772160);              // 128 MB  [E*C][H] bf16
  float* eo   = (float*)(ws + 301989888);            //  64 MB  [E*C][M] f32
  int*   topk = (int*)(ws + 369098752);              //  64 KB
  float* gts  = (float*)(ws + 369098752 + 65536);    //  64 KB
  int*   flat = (int*)(ws + 369098752 + 131072);     //  64 KB
  int*   rsrc = (int*)(ws + 369098752 + 196608);     //  64 KB

  // 1) bf16 weight transposes (one-shot; weights then live in L2 for GEMMs)
  conv_w1_kernel<<<(NEXP * HDIM * MDIM) / 256, 256, 0, stream>>>(w1, w1t);
  conv_w2_kernel<<<(NEXP * MDIM * HDIM) / 256, 256, 0, stream>>>(w2, w2t);

  // 2) gating (wave per token)
  gate_kernel<<<S_TOK / 8, 256, 0, stream>>>(x, wg, topk, gts);

  // 3) capacity scan (wave per expert)
  scan_kernel<<<1, 256, 0, stream>>>(topk, flat, rsrc);

  // 4) dispatch gather -> bf16
  disp_kernel<<<NEXP * CAP, 256, 0, stream>>>(x, rsrc, disp);

  // 5) expert GEMM1 (fused bias+ReLU, bf16 out):  h = relu(disp @ w1 + b1)
  dim3 g1(CAP / 128, HDIM / 128, NEXP);
  moe_gemm_kernel<true><<<g1, 256, 0, stream>>>(disp, w1t, b1, hbuf, HDIM, MDIM);

  // 6) expert GEMM2 (fused bias, f32 out):  eo = h @ w2 + b2
  dim3 g2(CAP / 128, MDIM / 128, NEXP);
  moe_gemm_kernel<false><<<g2, 256, 0, stream>>>(hbuf, w2t, b2, eo, MDIM, HDIM);

  // 7) combine with gates
  combine_kernel<<<S_TOK, 256, 0, stream>>>(eo, flat, gts, (float*)d_out);
}
